// Attention_18064632447038
// MI455X (gfx1250) — matrix operand order repaired and checked by reading, not compiled, not run on hardware
//
#include <hip/hip_runtime.h>
#include <hip/hip_bf16.h>
#include <math.h>

#define B_ 128
#define S_ 4096
#define H_ 128
#define CHUNK 64            // s-rows per block in the scores kernel
#define NCHUNK (S_ / CHUNK) // 64 chunks per batch row
#define LDSTRIDE 136        // 128 + 8 halves: rotates rows by 4 LDS banks -> conflict-free

typedef __bf16        v16bf __attribute__((ext_vector_type(16)));
typedef float         v8f   __attribute__((ext_vector_type(8)));
typedef unsigned int  v4u   __attribute__((ext_vector_type(4)));

__device__ __forceinline__ unsigned short f2bf(float x) {
    unsigned int u = __float_as_uint(x);
    unsigned int r = u + 0x7FFFu + ((u >> 16) & 1u);   // round-to-nearest-even
    return (unsigned short)(r >> 16);
}

// ---------------- Kernel 0: h_proj, We->bf16, zero weighted output -----------
__global__ __launch_bounds__(128)
void prep_kernel(const float* __restrict__ hidden, const float* __restrict__ attn_w,
                 const float* __restrict__ attn_b, float* __restrict__ hproj,
                 unsigned short* __restrict__ we_bf, float* __restrict__ out_weighted) {
    const int blk = blockIdx.x;    // doubles as b (h_proj) and g (We row)
    const int tid = threadIdx.x;   // doubles as g (h_proj) and h (We col)
    __shared__ float hrow[H_];
    hrow[tid] = hidden[blk * H_ + tid];
    __syncthreads();
    float acc = attn_b[tid];
    const float* wrow = attn_w + (size_t)tid * (2 * H_);
    #pragma unroll 8
    for (int h = 0; h < H_; ++h) acc = fmaf(hrow[h], wrow[h], acc);
    hproj[blk * H_ + tid] = acc;
    we_bf[blk * H_ + tid] = f2bf(attn_w[(size_t)blk * (2 * H_) + H_ + tid]);
    out_weighted[blk * H_ + tid] = 0.0f;   // needed by fallback path; harmless otherwise
}

// ---------------- Kernel 1: scores via bf16 WMMA (+ optional online-softmax partials)
// grid: (NCHUNK, B), block: 128 threads = 4 waves; wave w -> 16 s-rows.
__global__ __launch_bounds__(128)
void scores_kernel(const float* __restrict__ enc, const float* __restrict__ hproj,
                   const unsigned short* __restrict__ we_bf, const float* __restrict__ v_w,
                   float* __restrict__ scores,
                   float* __restrict__ stats,      // [B*NCHUNK][2] : (m_c, l_c)
                   float* __restrict__ partial,    // [B*NCHUNK][H] : partial weighted
                   int fused) {
    const int b      = blockIdx.y;
    const int chunk  = blockIdx.x;
    const int s_base = chunk * CHUNK;
    const int tid    = threadIdx.x;
    const int lane   = tid & 31;
    const int wave   = tid >> 5;
    const int row    = lane & 15;
    const int hs     = lane >> 4;        // half-wave select

    __shared__ unsigned short lds_enc[CHUNK * LDSTRIDE];
    __shared__ unsigned short lds_we [H_ * LDSTRIDE];
    __shared__ float sc[CHUNK];

    // --- stage We (bf16) into padded LDS: 4096 chunks of 4 halves ---
    const uint2* wsrc = (const uint2*)we_bf;
    #pragma unroll
    for (int i = 0; i < 32; ++i) {
        int idx = tid + i * 128;
        int g = idx >> 5, hc = (idx & 31) << 2;
        *(uint2*)&lds_we[g * LDSTRIDE + hc] = wsrc[idx];
    }
    // --- stage enc tile (64 x 128 f32 -> bf16): 2048 chunks of 4 ---
    const float4* esrc = (const float4*)(enc + ((size_t)b * S_ + s_base) * H_);
    #pragma unroll
    for (int i = 0; i < 16; ++i) {
        int idx = tid + i * 128;
        int sl = idx >> 5, hc = (idx & 31) << 2;
        float4 v = esrc[idx];
        unsigned int lo = (unsigned int)f2bf(v.x) | ((unsigned int)f2bf(v.y) << 16);
        unsigned int hi = (unsigned int)f2bf(v.z) | ((unsigned int)f2bf(v.w) << 16);
        *(uint2*)&lds_enc[sl * LDSTRIDE + hc] = make_uint2(lo, hi);
    }
    __syncthreads();

    // --- e_proj tile: 16(s) x 128(g), K=H=128 in 4 chunks of 32 ---
    v8f acc[8] = {};
    const int arow = wave * 16 + row;
    #pragma unroll
    for (int c = 0; c < 4; ++c) {
        int h0 = c * 32 + hs * 8;
        v4u a0 = *(const v4u*)&lds_enc[arow * LDSTRIDE + h0];
        v4u a1 = *(const v4u*)&lds_enc[arow * LDSTRIDE + h0 + 16];
        v16bf a = __builtin_bit_cast(v16bf, __builtin_shufflevector(a0, a1, 0,1,2,3,4,5,6,7));
        #pragma unroll
        for (int t = 0; t < 8; ++t) {
            int g  = t * 16 + row;
            int hb = c * 32 + hs * 8;
            v4u b0 = *(const v4u*)&lds_we[g * LDSTRIDE + hb];
            v4u b1 = *(const v4u*)&lds_we[g * LDSTRIDE + hb + 16];
            v16bf bm = __builtin_bit_cast(v16bf, __builtin_shufflevector(b0, b1, 0,1,2,3,4,5,6,7));
            acc[t] = __builtin_amdgcn_wmma_f32_16x16x32_bf16(
                false, a, false, bm, (short)0, acc[t], false, false);
        }
    }

    // --- epilogue: tanh(hp + e)·v_w, reduce over g, write raw scores ---
    float hp[8], vw[8];
    #pragma unroll
    for (int t = 0; t < 8; ++t) {
        int g = t * 16 + row;
        hp[t] = hproj[b * H_ + g];
        vw[t] = v_w[g];
    }
    #pragma unroll
    for (int j = 0; j < 8; ++j) {
        float sum = 0.0f;
        #pragma unroll
        for (int t = 0; t < 8; ++t)
            sum += tanhf(acc[t][j] + hp[t]) * vw[t];
        sum += __shfl_xor(sum, 1);
        sum += __shfl_xor(sum, 2);
        sum += __shfl_xor(sum, 4);
        sum += __shfl_xor(sum, 8);
        if (row == 0) {
            int sl = wave * 16 + j + hs * 8;
            scores[(size_t)b * S_ + s_base + sl] = sum;   // raw scores (pre-softmax)
            sc[sl] = sum;
        }
    }

    if (!fused) return;

    // --- online-softmax chunk stats + partial weighted sum (enc rows hot in L2) ---
    __syncthreads();
    float m_c = -INFINITY;
    for (int s = 0; s < CHUNK; ++s) m_c = fmaxf(m_c, sc[s]);   // LDS broadcast reads
    __syncthreads();
    if (tid < CHUNK) sc[tid] = expf(sc[tid] - m_c);
    __syncthreads();
    if (tid == 0) {
        float l_c = 0.0f;
        for (int s = 0; s < CHUNK; ++s) l_c += sc[s];
        stats[((size_t)b * NCHUNK + chunk) * 2 + 0] = m_c;
        stats[((size_t)b * NCHUNK + chunk) * 2 + 1] = l_c;
    }
    // thread = h column; f32 enc re-read hits WGP$/L2 (just loaded above)
    {
        const float* ecol = enc + ((size_t)b * S_ + s_base) * H_ + tid;
        float wacc = 0.0f;
        #pragma unroll 4
        for (int s = 0; s < CHUNK; ++s) wacc = fmaf(sc[s], ecol[(size_t)s * H_], wacc);
        partial[((size_t)b * NCHUNK + chunk) * H_ + tid] = wacc;
    }
}

// ---------------- Kernel 2 (fused path): combine chunks, normalize -----------
__global__ __launch_bounds__(128)
void combine_kernel(float* __restrict__ attn,            // in: raw scores, out: softmax
                    const float* __restrict__ stats,
                    const float* __restrict__ partial,
                    float* __restrict__ out_weighted) {
    const int b = blockIdx.x, tid = threadIdx.x;
    __shared__ float cm[NCHUNK], cf[NCHUNK];
    if (tid < NCHUNK) {
        cm[tid] = stats[((size_t)b * NCHUNK + tid) * 2 + 0];
        cf[tid] = stats[((size_t)b * NCHUNK + tid) * 2 + 1];
    }
    __syncthreads();
    float M = -INFINITY;
    for (int c = 0; c < NCHUNK; ++c) M = fmaxf(M, cm[c]);
    __syncthreads();
    if (tid < NCHUNK) {
        float coef = expf(cm[tid] - M);
        cm[tid] = coef;              // reuse cm[] as rescale coefficients
        cf[tid] = cf[tid] * coef;    // rescaled partial sums
    }
    __syncthreads();
    float L = 0.0f;
    for (int c = 0; c < NCHUNK; ++c) L += cf[c];
    const float invL = 1.0f / L;
    // weighted[b, h=tid] = (1/L) * sum_c coef_c * partial_c[h]
    float wacc = 0.0f;
    #pragma unroll 4
    for (int c = 0; c < NCHUNK; ++c)
        wacc = fmaf(cm[c], partial[((size_t)b * NCHUNK + c) * H_ + tid], wacc);
    out_weighted[b * H_ + tid] = wacc * invL;
    // normalize attention weights in place: 4096 elems, 32 per thread
    float* rowp = attn + (size_t)b * S_;
    #pragma unroll 4
    for (int i = 0; i < S_ / 128; ++i) {
        int s = tid + i * 128;
        rowp[s] = expf(rowp[s] - M) * invL;
    }
}

// ---------------- Fallback kernels (two-pass, small workspace) ---------------
__global__ __launch_bounds__(256)
void softmax_kernel(float* __restrict__ attn) {
    const int b = blockIdx.x, tid = threadIdx.x;
    float* rowp = attn + (size_t)b * S_;
    __shared__ float red[256];
    float vals[16];
    float m = -INFINITY;
    #pragma unroll
    for (int i = 0; i < 16; ++i) { vals[i] = rowp[tid + i * 256]; m = fmaxf(m, vals[i]); }
    red[tid] = m; __syncthreads();
    for (int s = 128; s > 0; s >>= 1) {
        if (tid < s) red[tid] = fmaxf(red[tid], red[tid + s]);
        __syncthreads();
    }
    m = red[0]; __syncthreads();
    float sum = 0.0f;
    #pragma unroll
    for (int i = 0; i < 16; ++i) { vals[i] = expf(vals[i] - m); sum += vals[i]; }
    red[tid] = sum; __syncthreads();
    for (int s = 128; s > 0; s >>= 1) {
        if (tid < s) red[tid] += red[tid + s];
        __syncthreads();
    }
    const float inv = 1.0f / red[0];
    #pragma unroll
    for (int i = 0; i < 16; ++i) rowp[tid + i * 256] = vals[i] * inv;
}

__global__ __launch_bounds__(128)
void weighted_kernel(const float* __restrict__ enc, const float* __restrict__ attn,
                     float* __restrict__ out_weighted) {
    const int b = blockIdx.y, chunk = blockIdx.x, h = threadIdx.x;
    const float* e = enc + ((size_t)b * S_ + (size_t)chunk * 512) * H_ + h;
    const float* w = attn + (size_t)b * S_ + (size_t)chunk * 512;
    float acc = 0.0f;
    #pragma unroll 4
    for (int s = 0; s < 512; ++s) acc = fmaf(w[s], e[(size_t)s * H_], acc);
    atomicAdd(&out_weighted[b * H_ + h], acc);
}

extern "C" void kernel_launch(void* const* d_in, const int* in_sizes, int n_in,
                              void* d_out, int out_size, void* d_ws, size_t ws_size,
                              hipStream_t stream) {
    const float* hidden = (const float*)d_in[0];
    const float* enc    = (const float*)d_in[1];
    const float* attn_w = (const float*)d_in[2];
    const float* attn_b = (const float*)d_in[3];
    const float* v_w    = (const float*)d_in[4];

    float* out_weighted = (float*)d_out;                 // [B,H]
    float* out_attn     = (float*)d_out + B_ * H_;       // [B,S]

    // workspace layout
    char* ws = (char*)d_ws;
    float*          hproj  = (float*)ws;                           // 64 KB
    unsigned short* we_bf  = (unsigned short*)(ws + 65536);        // 32 KB
    float*          stats  = (float*)(ws + 98304);                 // 64 KB
    float*          part   = (float*)(ws + 163840);                // 4 MB
    const size_t need_fused = 163840 + (size_t)B_ * NCHUNK * H_ * sizeof(float);

    prep_kernel<<<B_, H_, 0, stream>>>(hidden, attn_w, attn_b, hproj, we_bf, out_weighted);

    if (ws_size >= need_fused) {
        // single HBM pass over encoder_outputs (online softmax)
        scores_kernel<<<dim3(NCHUNK, B_), 128, 0, stream>>>(
            enc, hproj, we_bf, v_w, out_attn, stats, part, 1);
        combine_kernel<<<B_, 128, 0, stream>>>(out_attn, stats, part, out_weighted);
    } else {
        // two-pass fallback (minimal workspace)
        scores_kernel<<<dim3(NCHUNK, B_), 128, 0, stream>>>(
            enc, hproj, we_bf, v_w, out_attn, nullptr, nullptr, 0);
        softmax_kernel<<<B_, 256, 0, stream>>>(out_attn);
        weighted_kernel<<<dim3(8, B_), 128, 0, stream>>>(enc, out_attn, out_weighted);
    }
}
